// _UnidirectionalAttention_31207232372831
// MI455X (gfx1250) — matrix operand order repaired and checked by reading, not compiled, not run on hardware
//
#include <hip/hip_runtime.h>
#include <hip/hip_bf16.h>
#include <math.h>
#include <stdint.h>

// ---- problem constants (B=1, N=4096, D=1024, H=16, HS=64) ----
#define SEQ   4096
#define DIM   1024
#define NHEAD 16
#define HSZ   64
#define NEG_BIG (-10000.0f)

typedef __attribute__((ext_vector_type(16))) __bf16 v16bf;
typedef __attribute__((ext_vector_type(8)))  __bf16 v8bf;
typedef __attribute__((ext_vector_type(8)))  float  v8f;
typedef __attribute__((ext_vector_type(4)))  float  v4f;
typedef __attribute__((ext_vector_type(4)))  unsigned int v4u;
typedef __attribute__((ext_vector_type(4)))  int v4i;
typedef __attribute__((ext_vector_type(8)))  int v8i;

__device__ __forceinline__ v16bf cat16(v8bf lo, v8bf hi) {
  union { v16bf v; v8bf h[2]; } u;
  u.h[0] = lo; u.h[1] = hi;
  return u.v;
}

__device__ __forceinline__ v8f wmma_bf16(v16bf a, v16bf b, v8f c) {
  return __builtin_amdgcn_wmma_f32_16x16x32_bf16(
      false, a, false, b, (short)0, c, false, false);
}

// ---- DPP row_ror reductions across 16-lane rows (stays within each half) ---
#define DPP_ROR_F(x, n)                                                   \
  __int_as_float(__builtin_amdgcn_update_dpp(                             \
      0, __float_as_int(x), 0x120 + (n), 0xF, 0xF, true))

__device__ __forceinline__ float red16_max(float x) {
  x = fmaxf(x, DPP_ROR_F(x, 1));
  x = fmaxf(x, DPP_ROR_F(x, 2));
  x = fmaxf(x, DPP_ROR_F(x, 4));
  x = fmaxf(x, DPP_ROR_F(x, 8));
  return x;
}
__device__ __forceinline__ float red16_sum(float x) {
  x += DPP_ROR_F(x, 1);
  x += DPP_ROR_F(x, 2);
  x += DPP_ROR_F(x, 4);
  x += DPP_ROR_F(x, 8);
  return x;
}

// ============================================================================
// Tensor Data Mover: 2D tile load (bf16 elements) global -> LDS.
// D# packed per CDNA5 ISA.
// ============================================================================
__device__ __forceinline__ void tdm_load_2d(
    unsigned lds_off, const void* gp,
    unsigned tile_d0, unsigned tile_d1,
    unsigned tens_d0, unsigned tens_d1,
    unsigned long long stride0_elems) {
  unsigned long long ga = (unsigned long long)(uintptr_t)gp;
  v4u g0;
  g0[0] = 1u;                                   // count=1 (valid user D#)
  g0[1] = lds_off;                              // lds_addr (bytes)
  g0[2] = (unsigned)(ga & 0xffffffffu);         // global_addr[31:0]
  g0[3] = (unsigned)((ga >> 32) & 0x1ffffffu)   // global_addr[56:32]
        | (2u << 30);                           // type=2 ("image")
  v8i g1;
  g1[0] = (int)(1u << 16);                      // data_size=1 -> 2 bytes
  g1[1] = (int)((tens_d0 & 0xffffu) << 16);     // tensor_dim0[15:0]
  g1[2] = (int)(((tens_d0 >> 16) & 0xffffu) | ((tens_d1 & 0xffffu) << 16));
  g1[3] = (int)(((tens_d1 >> 16) & 0xffffu) | ((tile_d0 & 0xffffu) << 16));
  g1[4] = (int)(tile_d1 & 0xffffu);             // tile_dim1 (tile_dim2=0)
  g1[5] = (int)(stride0_elems & 0xffffffffu);   // tensor_dim0_stride[31:0]
  g1[6] = (int)((stride0_elems >> 32) & 0xffffu);
  g1[7] = 0;
  v4i gz = {0, 0, 0, 0};
#if __clang_major__ >= 23
  v8i gz8 = {0, 0, 0, 0, 0, 0, 0, 0};
  __builtin_amdgcn_tensor_load_to_lds(g0, g1, gz, gz, gz8, 0);
#else
  __builtin_amdgcn_tensor_load_to_lds(g0, g1, gz, gz, 0);
#endif
}

// ============================================================================
// GEMM: 64x64 tile = A[M,K] x B[K,Nc] (+bias), bf16 WMMA, f32 accumulate.
// EPI==0: QKV epilogue -> per-head Q(scaled)/K bf16 [H][N][HS], V transposed
//         bf16 [H][HS][N].  EPI==1: f32 store to d_out.
// ============================================================================
template <int EPI, bool A_BF16>
__global__ __launch_bounds__(128) void gemm64(
    const void* __restrict__ Av, const float* __restrict__ Bw,
    const float* __restrict__ bias, int K, int Nc,
    __bf16* __restrict__ Qd, __bf16* __restrict__ Kd, __bf16* __restrict__ Vt,
    float* __restrict__ Out) {
  __shared__ __align__(16) __bf16 As[64][40];  // [m][k], padded rows
  __shared__ __align__(16) __bf16 Bs[64][40];  // [n][k] (transposed), padded

  const int tid = threadIdx.x;
  const int m0 = blockIdx.y * 64;
  const int n0 = blockIdx.x * 64;
  const int w = tid >> 5, lane = tid & 31;
  const int lh = lane >> 4, ln = lane & 15;
  const int wm = (w >> 1) * 32, wn = (w & 1) * 32;

  v8f acc[2][2] = {};

  for (int k0 = 0; k0 < K; k0 += 32) {
    __syncthreads();
    // ---- stage A tile (64 x 32) into LDS as bf16 ----
    if (A_BF16) {
      const __bf16* A = (const __bf16*)Av;
#pragma unroll
      for (int it = 0; it < 2; ++it) {
        int i = tid + it * 128;
        int row = i >> 2, cq = i & 3;
        v8bf v = *(const v8bf*)(A + (size_t)(m0 + row) * K + k0 + cq * 8);
        *(v8bf*)&As[row][cq * 8] = v;
        if (k0 + 32 < K)
          __builtin_prefetch(A + (size_t)(m0 + row) * K + k0 + 32 + cq * 8, 0, 0);
      }
    } else {
      const float* A = (const float*)Av;
#pragma unroll
      for (int it = 0; it < 4; ++it) {
        int i = tid + it * 128;
        int row = i >> 3, cq = i & 7;
        v4f v = *(const v4f*)(A + (size_t)(m0 + row) * K + k0 + cq * 4);
#pragma unroll
        for (int j = 0; j < 4; ++j) As[row][cq * 4 + j] = (__bf16)v[j];
        if (k0 + 32 < K)
          __builtin_prefetch(A + (size_t)(m0 + row) * K + k0 + 32 + cq * 4, 0, 0);
      }
    }
    // ---- stage B tile (32 x 64), transposed into Bs[n][k] as bf16 ----
#pragma unroll
    for (int it = 0; it < 4; ++it) {
      int i = tid + it * 128;
      int row = i >> 4, cq = i & 15;
      v4f v = *(const v4f*)(Bw + (size_t)(k0 + row) * Nc + n0 + cq * 4);
#pragma unroll
      for (int j = 0; j < 4; ++j) Bs[cq * 4 + j][row] = (__bf16)v[j];
      if (k0 + 32 < K)
        __builtin_prefetch(Bw + (size_t)(k0 + 32 + row) * Nc + n0 + cq * 4, 0, 0);
    }
    __syncthreads();

    // A frag: lane M=ln, K = e + 8*lh (+8 for e>=8).  B frag: lane N=ln,
    // K = e + 8*lh (+8 for e>=8).  All 4 fragments loaded up front (batched b128 ds
    // loads), then the 4 WMMAs issue back-to-back.
    v16bf af[2], bfg[2];
#pragma unroll
    for (int sm = 0; sm < 2; ++sm) {
      int r = wm + sm * 16 + ln;
      af[sm] = cat16(*(const v8bf*)&As[r][lh * 8],
                     *(const v8bf*)&As[r][lh * 8 + 16]);
    }
#pragma unroll
    for (int sn = 0; sn < 2; ++sn) {
      int r = wn + sn * 16 + ln;
      bfg[sn] = cat16(*(const v8bf*)&Bs[r][lh * 8],
                      *(const v8bf*)&Bs[r][lh * 8 + 16]);
    }
#pragma unroll
    for (int sm = 0; sm < 2; ++sm)
#pragma unroll
      for (int sn = 0; sn < 2; ++sn)
        acc[sm][sn] = wmma_bf16(af[sm], bfg[sn], acc[sm][sn]);
  }

  // ---- epilogue: D element (r,lane) -> row M = r + 8*lh, col N = ln ----
#pragma unroll
  for (int sm = 0; sm < 2; ++sm) {
#pragma unroll
    for (int sn = 0; sn < 2; ++sn) {
      int gc = n0 + wn + sn * 16 + ln;
      float bv = bias[gc];
#pragma unroll
      for (int r = 0; r < 8; ++r) {
        int gm = m0 + wm + sm * 16 + r + lh * 8;
        float val = acc[sm][sn][r] + bv;
        if (EPI == 0) {
          int hd = gc / 192, l = gc % 192;
          if (l < 64) {
            Qd[((size_t)hd * SEQ + gm) * HSZ + l] = (__bf16)(val * 0.125f);
          } else if (l < 128) {
            Kd[((size_t)hd * SEQ + gm) * HSZ + (l - 64)] = (__bf16)val;
          } else {  // V stored transposed: Vt[h][hs][n]
            Vt[((size_t)hd * HSZ + (l - 128)) * SEQ + gm] = (__bf16)val;
          }
        } else {
          Out[(size_t)gm * Nc + gc] = val;
        }
      }
    }
  }
}

// ============================================================================
// Flash attention: block = (64 queries, head); 4 waves x 16 queries.
// K and V^T tiles staged by the Tensor Data Mover, double buffered, so the
// DMA of tile t+1 overlaps the WMMAs of tile t (s_wait_tensorcnt pipeline).
// ============================================================================
__global__ __launch_bounds__(128) void attn_kernel(
    const __bf16* __restrict__ Qd, const __bf16* __restrict__ Kd,
    const __bf16* __restrict__ Vt, __bf16* __restrict__ Cmb) {
  __shared__ __align__(16) __bf16 Ks[2][32][64];   // [key][hs], dbl-buffered
  __shared__ __align__(16) __bf16 Vts[2][64][32];  // [hs][key], dbl-buffered
  __shared__ __align__(16) __bf16 Ps[4][16][40];   // per-wave P tile [q][key]

  const int tid = threadIdx.x;
  const int hd = blockIdx.y;
  const int qblk = blockIdx.x;
  const int w = tid >> 5, lane = tid & 31;
  const int lh = lane >> 4, ln = lane & 15;
  const int qb = qblk * 64 + w * 16;

  const __bf16* Qh = Qd + (size_t)hd * SEQ * HSZ;
  const __bf16* Kh = Kd + (size_t)hd * SEQ * HSZ;
  const __bf16* Vh = Vt + (size_t)hd * HSZ * SEQ;  // [HS][N]

  // Q A-fragments for hs chunks [0,32) and [32,64); 1/8 scale baked in.
  v16bf qf[2];
  {
    const __bf16* qrow = Qh + (size_t)(qb + ln) * HSZ;
#pragma unroll
    for (int c = 0; c < 2; ++c)
      qf[c] = cat16(*(const v8bf*)(qrow + c * 32 + lh * 8),
                    *(const v8bf*)(qrow + c * 32 + lh * 8 + 16));
  }

  v8f accO[4] = {};
  float mrow[8], lrow[8];
#pragma unroll
  for (int r = 0; r < 8; ++r) { mrow[r] = -1e30f; lrow[r] = 0.0f; }

  const int nkt = qblk * 2 + 2;  // 32-key tiles covering keys <= max query

  auto issue_tiles = [&](int kt, int b) {
    const int k0 = kt * 32;
    tdm_load_2d((unsigned)(uintptr_t)&Ks[b][0][0], Kh + (size_t)k0 * HSZ,
                /*tile*/ HSZ, 32, /*tensor*/ HSZ, SEQ, /*stride0*/ HSZ);
    tdm_load_2d((unsigned)(uintptr_t)&Vts[b][0][0], Vh + k0,
                /*tile*/ 32, HSZ, /*tensor*/ SEQ, HSZ, /*stride0*/ SEQ);
  };

  if (w == 0) issue_tiles(0, 0);

  for (int kt = 0; kt < nkt; ++kt) {
    const int b = kt & 1;
    if (kt + 1 < nkt) {
      if (w == 0) issue_tiles(kt + 1, b ^ 1);   // prefetch next tile pair
      __builtin_amdgcn_s_wait_tensorcnt(2);     // tile kt landed in LDS
    } else {
      __builtin_amdgcn_s_wait_tensorcnt(0);
    }
    __syncthreads();  // publish buffer b

    // ---- load ALL K fragments first (8x ds_load_b128, clause-able),
    //      then issue the 4 S WMMAs back-to-back ----
    const int k0 = kt * 32;
    v16bf kf[2][2];
#pragma unroll
    for (int sk = 0; sk < 2; ++sk) {
      const __bf16* krow = &Ks[b][sk * 16 + ln][0];
#pragma unroll
      for (int c = 0; c < 2; ++c)
        kf[sk][c] = cat16(*(const v8bf*)(krow + c * 32 + lh * 8),
                          *(const v8bf*)(krow + c * 32 + lh * 8 + 16));
    }
    v8f S[2];
#pragma unroll
    for (int sk = 0; sk < 2; ++sk) {
      v8f s = {};
      s = wmma_bf16(qf[0], kf[sk][0], s);
      s = wmma_bf16(qf[1], kf[sk][1], s);
      S[sk] = s;
    }

    // ---- causal mask + online softmax; DPP row_ror lane reductions ----
#pragma unroll
    for (int r = 0; r < 8; ++r) {
      int q = qb + r + lh * 8;
      float tm = -1e30f;
#pragma unroll
      for (int sk = 0; sk < 2; ++sk) {
        int key = k0 + sk * 16 + ln;
        float v = (key <= q) ? S[sk][r] : NEG_BIG;
        S[sk][r] = v;
        tm = fmaxf(tm, v);
      }
      tm = red16_max(tm);
      float mnew = fmaxf(mrow[r], tm);
      float alpha = __expf(mrow[r] - mnew);
      float ls = 0.0f;
#pragma unroll
      for (int sk = 0; sk < 2; ++sk) {
        float p = __expf(S[sk][r] - mnew);
        S[sk][r] = p;
        ls += p;
      }
      ls = red16_sum(ls);
      lrow[r] = lrow[r] * alpha + ls;
      mrow[r] = mnew;
#pragma unroll
      for (int t = 0; t < 4; ++t) accO[t][r] *= alpha;
    }

    // ---- transpose P into per-wave LDS (A-fragment layout source) ----
#pragma unroll
    for (int sk = 0; sk < 2; ++sk)
#pragma unroll
      for (int r = 0; r < 8; ++r)
        Ps[w][r + lh * 8][sk * 16 + ln] = (__bf16)S[sk][r];
    asm volatile("s_wait_dscnt 0" ::: "memory");  // wave-local P visibility

    // ---- load P fragment + ALL V fragments, then 4 PV WMMAs ----
    v16bf pf;
    {
      const __bf16* prow = &Ps[w][ln][0];
      pf = cat16(*(const v8bf*)(prow + lh * 8),
                 *(const v8bf*)(prow + lh * 8 + 16));
    }
    v16bf vf[4];
#pragma unroll
    for (int sn = 0; sn < 4; ++sn) {
      const __bf16* vrow = &Vts[b][sn * 16 + ln][0];
      vf[sn] = cat16(*(const v8bf*)(vrow + lh * 8),
                     *(const v8bf*)(vrow + lh * 8 + 16));
    }
#pragma unroll
    for (int sn = 0; sn < 4; ++sn)
      accO[sn] = wmma_bf16(pf, vf[sn], accO[sn]);

    __syncthreads();  // all waves done with buffer b before DMA reuses it
  }

  // ---- finalize: combined[q][h*64+hs] = accO / l  (bf16) ----
#pragma unroll
  for (int sn = 0; sn < 4; ++sn) {
#pragma unroll
    for (int r = 0; r < 8; ++r) {
      int q = qb + r + lh * 8;
      int d = hd * HSZ + sn * 16 + ln;
      Cmb[(size_t)q * DIM + d] = (__bf16)(accO[sn][r] / lrow[r]);
    }
  }
}

// ============================================================================
extern "C" void kernel_launch(void* const* d_in, const int* in_sizes, int n_in,
                              void* d_out, int out_size, void* d_ws,
                              size_t ws_size, hipStream_t stream) {
  (void)in_sizes; (void)n_in; (void)out_size; (void)ws_size;
  const float* enc    = (const float*)d_in[0];
  const float* attn_W = (const float*)d_in[1];
  const float* attn_b = (const float*)d_in[2];
  const float* out_W  = (const float*)d_in[3];
  const float* out_b  = (const float*)d_in[4];
  float* out = (float*)d_out;

  char* ws = (char*)d_ws;
  const size_t MB8 = (size_t)8 * 1024 * 1024;
  __bf16* Qd  = (__bf16*)(ws);             // [H][N][HS] bf16, Q pre-scaled
  __bf16* Kd  = (__bf16*)(ws + 1 * MB8);   // [H][N][HS] bf16
  __bf16* Vtd = (__bf16*)(ws + 2 * MB8);   // [H][HS][N] bf16 (transposed)
  __bf16* Cmb = (__bf16*)(ws + 3 * MB8);   // [N][D] bf16 attention output

  dim3 blk(128);
  // 1) QKV projection: [4096,1024] x [1024,3072] + b -> Q/K/V^T
  gemm64<0, false><<<dim3(48, 64), blk, 0, stream>>>(
      (const void*)enc, attn_W, attn_b, DIM, 3 * DIM, Qd, Kd, Vtd, nullptr);
  // 2) causal flash attention, TDM-pipelined K/V^T tiles
  attn_kernel<<<dim3(SEQ / 64, NHEAD), blk, 0, stream>>>(Qd, Kd, Vtd, Cmb);
  // 3) output projection: [4096,1024] x [1024,1024] + b -> f32 out
  gemm64<1, true><<<dim3(16, 64), blk, 0, stream>>>(
      (const void*)Cmb, out_W, out_b, DIM, DIM, nullptr, nullptr, nullptr, out);
}
